// Attention_37726992728222
// MI455X (gfx1250) — compile-verified, matrix operand order checked by reading, not run on hardware
//
#include <hip/hip_runtime.h>

typedef __attribute__((ext_vector_type(16))) __bf16 v16bf;
typedef __attribute__((ext_vector_type(8)))  __bf16 v8bf;
typedef __attribute__((ext_vector_type(8)))  float  v8f;

union FragBF { v16bf f; v8bf h[2]; };

// ---------------------------------------------------------------------------
// Tensor Data Mover support (gfx1250). 2D tile load Global->LDS with LDS row
// padding so a 64-elem bf16 row lands at stride 72 halfs (144B, 16B aligned).
// ---------------------------------------------------------------------------
#if defined(__gfx1250__) && __has_builtin(__builtin_amdgcn_tensor_load_to_lds) && \
    __has_builtin(__builtin_amdgcn_s_wait_tensorcnt)
#define USE_TDM 1
#else
#define USE_TDM 0
#endif

#if USE_TDM
typedef __attribute__((ext_vector_type(4))) unsigned tdmv4u;
typedef __attribute__((ext_vector_type(8))) int      tdmv8i;
typedef __attribute__((ext_vector_type(4))) int      tdmv4i;

__device__ inline void tdm_set(unsigned* w, int lo, int len, unsigned long long v) {
  if (len < 64) v &= (1ull << len) - 1ull;
  const int word = lo >> 5, bit = lo & 31;
  w[word] |= (unsigned)(v << bit);
  if (bit + len > 32) w[word + 1] |= (unsigned)(v >> (32 - bit));
  if (bit + len > 64) w[word + 2] |= (unsigned)(v >> (64 - bit));
}

// Load a 128-row x 64-elem bf16 tile (row stride = rowStride elems in memory)
// into LDS at `ldsdst`, padded to 72-half rows. Issued per-wave; TENSORcnt.
__device__ inline void tdm_load_64x128(const void* gsrc, const void* ldsdst,
                                       unsigned rowStride) {
  unsigned g0w[4] = {0, 0, 0, 0};
  unsigned g1w[8] = {0, 0, 0, 0, 0, 0, 0, 0};
  tdm_set(g0w, 0, 2, 1);                                   // count = 1 (valid)
  tdm_set(g0w, 32, 32, (unsigned)(size_t)ldsdst);          // lds_addr (bytes)
  tdm_set(g0w, 64, 57, (unsigned long long)(size_t)gsrc);  // global_addr
  tdm_set(g0w, 126, 2, 2);                                 // type = 2 (image)

  tdm_set(g1w, 16, 2, 1);       // data_size = 2 bytes
  tdm_set(g1w, 20, 1, 1);       // pad_enable
  tdm_set(g1w, 22, 3, 4);       // pad_interval: 32 DWORDs (= one 64-bf16 row)
  tdm_set(g1w, 25, 7, 3);       // pad_amount: 4 DWORDs (8 halfs -> stride 72)
  tdm_set(g1w, 48, 32, 64);     // tensor_dim0
  tdm_set(g1w, 80, 32, 128);    // tensor_dim1
  tdm_set(g1w, 112, 16, 64);    // tile_dim0
  tdm_set(g1w, 128, 16, 128);   // tile_dim1 (tile_dim2 = 0 -> 2D)
  tdm_set(g1w, 160, 48, rowStride);  // tensor_dim0_stride (elems)

  tdmv4u g0; tdmv8i g1; tdmv4i gz;
  #pragma unroll
  for (int i = 0; i < 4; ++i) g0[i] = g0w[i];
  #pragma unroll
  for (int i = 0; i < 8; ++i) g1[i] = (int)g1w[i];
  #pragma unroll
  for (int i = 0; i < 4; ++i) gz[i] = 0;
#if __clang_major__ >= 23
  tdmv8i gz8;
  #pragma unroll
  for (int i = 0; i < 8; ++i) gz8[i] = 0;
  __builtin_amdgcn_tensor_load_to_lds(g0, g1, gz, gz, gz8, 0);
#else
  __builtin_amdgcn_tensor_load_to_lds(g0, g1, gz, gz, 0);
#endif
}
#endif  // USE_TDM

// ---------------------------------------------------------------------------
// 16x32 bf16 A-operand fragment (B-operand = A-layout of B^T).
// lane<16: row=lane, K={0..7,16..23}; lane>=16: row=lane-16, K={8..15,24..31}.
// stride must be a multiple of 8 halfs (16B-aligned b128 accesses).
// ---------------------------------------------------------------------------
__device__ inline v16bf ld_frag(const __bf16* base, int stride, int row16, int kbase) {
  const int lane = threadIdx.x & 31;
  const int row  = row16 + (lane & 15);
  const int k    = kbase + ((lane >> 4) << 3);
  const __bf16* p = base + row * stride + k;
  FragBF fr;
  fr.h[0] = *(const v8bf*)(p);
  fr.h[1] = *(const v8bf*)(p + 16);
  return fr.f;
}

// ---------------------------------------------------------------------------
// fp32 -> bf16 conversion (vectorized x4; all sizes are multiples of 4)
// ---------------------------------------------------------------------------
__global__ __launch_bounds__(256) void cvt_f32_to_bf16(const float* __restrict__ in,
                                                       __bf16* __restrict__ out, int n) {
  int i = (blockIdx.x * blockDim.x + threadIdx.x) * 4;
  if (i < n) {
    float4 v = *(const float4*)(in + i);
    out[i + 0] = (__bf16)v.x;
    out[i + 1] = (__bf16)v.y;
    out[i + 2] = (__bf16)v.z;
    out[i + 3] = (__bf16)v.w;
  }
}

// ---------------------------------------------------------------------------
// C[M,N] = A[M,K] * Bw[N,K]^T (+bias). BM=BN=128, BK=64, 8 waves,
// each wave computes a 32x64 tile = 2x4 wmma tiles, fp32 accumulation.
// TDM path: double-buffered LDS, wave 0 posts next slab while block computes.
// ---------------------------------------------------------------------------
template <bool OUT_BF16, bool BIAS>
__global__ __launch_bounds__(256)
void gemm_wmma(const __bf16* __restrict__ A, const __bf16* __restrict__ Bw,
               const float* __restrict__ bias, void* __restrict__ Cout,
               int K, int ldc) {
  constexpr int BM = 128, BN = 128, BK = 64, STR = BK + 8;
#if USE_TDM
  constexpr int NBUF = 2;
#else
  constexpr int NBUF = 1;
#endif
  __shared__ __bf16 As[NBUF][BM * STR];
  __shared__ __bf16 Bs[NBUF][BN * STR];

  const int tid  = threadIdx.x;
  const int wave = tid >> 5;
  const int lane = tid & 31;
  const int m0 = blockIdx.y * BM;
  const int n0 = blockIdx.x * BN;
  const int wm = (wave >> 1) * 32;
  const int wn = (wave & 1) * 64;

  v8f acc[2][4] = {};

  auto compute_slab = [&](const __bf16* Asl, const __bf16* Bsl) {
    #pragma unroll
    for (int ks = 0; ks < 2; ++ks) {
      v16bf af[2], bf[4];
      #pragma unroll
      for (int i = 0; i < 2; ++i) af[i] = ld_frag(Asl, STR, wm + i * 16, ks * 32);
      #pragma unroll
      for (int j = 0; j < 4; ++j) bf[j] = ld_frag(Bsl, STR, wn + j * 16, ks * 32);
      #pragma unroll
      for (int i = 0; i < 2; ++i)
        #pragma unroll
        for (int j = 0; j < 4; ++j)
          acc[i][j] = __builtin_amdgcn_wmma_f32_16x16x32_bf16(
              false, af[i], false, bf[j], (short)0, acc[i][j], false, false);
    }
  };

  const __bf16* Arow = A  + (size_t)m0 * K;
  const __bf16* Brow = Bw + (size_t)n0 * K;
  const int nIter = K / BK;

#if USE_TDM
  if (wave == 0) {
    tdm_load_64x128(Arow, As[0], (unsigned)K);
    tdm_load_64x128(Brow, Bs[0], (unsigned)K);
  }
  for (int i = 0; i < nIter; ++i) {
    const int cur = i & 1;
    if (wave == 0) {
      if (i + 1 < nIter) {
        // buffer cur^1 was last read before the previous barrier -> safe
        tdm_load_64x128(Arow + (size_t)(i + 1) * BK, As[cur ^ 1], (unsigned)K);
        tdm_load_64x128(Brow + (size_t)(i + 1) * BK, Bs[cur ^ 1], (unsigned)K);
        __builtin_amdgcn_s_wait_tensorcnt((short)2);  // slab i resident (in-order)
      } else {
        __builtin_amdgcn_s_wait_tensorcnt((short)0);
      }
    }
    __syncthreads();
    compute_slab(As[cur], Bs[cur]);
    __syncthreads();
  }
#else
  for (int i = 0; i < nIter; ++i) {
    const int k0 = i * BK;
    __syncthreads();
    #pragma unroll
    for (int c = 0; c < 4; ++c) {
      int chunk = tid + c * 256;
      int r  = chunk >> 3;
      int kk = (chunk & 7) << 3;
      *(v8bf*)&As[0][r * STR + kk] = *(const v8bf*)&Arow[(size_t)r * K + k0 + kk];
      *(v8bf*)&Bs[0][r * STR + kk] = *(const v8bf*)&Brow[(size_t)r * K + k0 + kk];
      if (k0 + BK < K) {
        __builtin_prefetch(&Arow[(size_t)r * K + k0 + BK + kk], 0, 3);
        __builtin_prefetch(&Brow[(size_t)r * K + k0 + BK + kk], 0, 3);
      }
    }
    __syncthreads();
    compute_slab(As[0], Bs[0]);
  }
#endif

  // C/D layout: VGPR r -> row r (lanes 0-15) / row r+8 (lanes 16-31), col=lane&15
  const int rbase = (lane >> 4) << 3;
  const int cidx  = lane & 15;
  #pragma unroll
  for (int i = 0; i < 2; ++i)
    #pragma unroll
    for (int j = 0; j < 4; ++j) {
      const int col = n0 + wn + j * 16 + cidx;
      const float bv = BIAS ? bias[col] : 0.0f;
      #pragma unroll
      for (int r = 0; r < 8; ++r) {
        const int row = m0 + wm + i * 16 + rbase + r;
        const float v = acc[i][j][r] + bv;
        if (OUT_BF16) ((__bf16*)Cout)[(size_t)row * ldc + col] = (__bf16)v;
        else          ((float*) Cout)[(size_t)row * ldc + col] = v;
      }
    }
}

// ---------------------------------------------------------------------------
// Flash-style causal attention. qkv: [B*N, 3072] bf16, head h at cols
// {h*64 | 1024+h*64 | 2048+h*64}. One block = 128 query rows of one (b,h);
// 8 waves, each wave owns 16 query rows. out: [B*N, 1024] bf16.
// K tile staged by TDM; V transposed into LDS by all threads in parallel.
// ---------------------------------------------------------------------------
__global__ __launch_bounds__(256)
void attn_wmma(const __bf16* __restrict__ qkv, __bf16* __restrict__ out) {
  constexpr int Nq = 2048, D = 64, Cq = 3072;
  constexpr int KSTR = 72, VSTR = 136, PSTR = 136;
  __shared__ __bf16 Ks[128 * KSTR];      // K block, row-major [key][d]
  __shared__ __bf16 Vt[64 * VSTR];       // V block transposed [d][key]
  __shared__ __bf16 Pb[8 * 16 * PSTR];   // per-wave P, [qrow][key]

  const int tid  = threadIdx.x;
  const int wave = tid >> 5;
  const int lane = tid & 31;
  const int qblk = blockIdx.x;
  const int bh   = blockIdx.y;
  const int b = bh >> 4, h = bh & 15;
  const int rowbase = b * Nq;
  const int qcol = h * D, kcol = 1024 + h * D, vcol = 2048 + h * D;
  const int q0 = qblk * 128 + wave * 16;
  const float scale = 0.125f;  // D^-0.5

  // Q fragments (A operand) straight from global; D=64 -> 2 frags
  v16bf qf[2];
  {
    const int row  = q0 + (lane & 15);
    const int koff = (lane >> 4) << 3;
    const __bf16* p0 = qkv + (size_t)(rowbase + row) * Cq + qcol;
    #pragma unroll
    for (int ks = 0; ks < 2; ++ks) {
      FragBF fr;
      fr.h[0] = *(const v8bf*)(p0 + ks * 32 + koff);
      fr.h[1] = *(const v8bf*)(p0 + ks * 32 + koff + 16);
      qf[ks] = fr.f;
    }
  }

  v8f o[4] = {};
  float mrow[8], lrow[8];
  #pragma unroll
  for (int r = 0; r < 8; ++r) { mrow[r] = -1e30f; lrow[r] = 0.0f; }

  __bf16* Pw = Pb + wave * 16 * PSTR;
  const int rbase = (lane >> 4) << 3;
  const int cidx  = lane & 15;

  for (int j = 0; j <= qblk; ++j) {
    const int kbase = j * 128;
    __syncthreads();  // previous K/V tiles fully consumed
#if USE_TDM
    if (wave == 0)
      tdm_load_64x128(&qkv[(size_t)(rowbase + kbase) * Cq + kcol], Ks, Cq);
#else
    #pragma unroll
    for (int c = 0; c < 4; ++c) {
      int chunk = tid + c * 256;
      int r  = chunk >> 3;
      int kk = (chunk & 7) << 3;
      *(v8bf*)&Ks[r * KSTR + kk] =
          *(const v8bf*)&qkv[(size_t)(rowbase + kbase + r) * Cq + kcol + kk];
    }
#endif
    #pragma unroll
    for (int c = 0; c < 4; ++c) {  // V tile, transposed into LDS
      int chunk = tid + c * 256;
      int kr = chunk >> 3;
      int d0 = (chunk & 7) << 3;
      v8bf vv = *(const v8bf*)&qkv[(size_t)(rowbase + kbase + kr) * Cq + vcol + d0];
      #pragma unroll
      for (int e = 0; e < 8; ++e) Vt[(d0 + e) * VSTR + kr] = vv[e];
    }
#if USE_TDM
    if (wave == 0) __builtin_amdgcn_s_wait_tensorcnt((short)0);
#endif
    __syncthreads();

    // S = Q K^T : 8 tiles of 16x16, 2 k-steps each
    v8f s[8];
    #pragma unroll
    for (int nt = 0; nt < 8; ++nt) {
      v8f a = {};
      #pragma unroll
      for (int ks = 0; ks < 2; ++ks) {
        v16bf kf = ld_frag(Ks, KSTR, nt * 16, ks * 32);
        a = __builtin_amdgcn_wmma_f32_16x16x32_bf16(false, qf[ks], false, kf,
                                                    (short)0, a, false, false);
      }
      s[nt] = a;
    }

    // scale + causal mask (diagonal block only)
    #pragma unroll
    for (int nt = 0; nt < 8; ++nt)
      #pragma unroll
      for (int r = 0; r < 8; ++r) {
        float v = s[nt][r] * scale;
        if (j == qblk) {
          int qi = q0 + rbase + r;
          int kj = kbase + nt * 16 + cidx;
          if (kj > qi) v = -1e30f;
        }
        s[nt][r] = v;
      }

    // online softmax; each row lives in a 16-lane half -> xor-shuffle reductions
    #pragma unroll
    for (int r = 0; r < 8; ++r) {
      float mx = s[0][r];
      #pragma unroll
      for (int nt = 1; nt < 8; ++nt) mx = fmaxf(mx, s[nt][r]);
      #pragma unroll
      for (int off = 1; off < 16; off <<= 1) mx = fmaxf(mx, __shfl_xor(mx, off, 32));
      const float mnew = fmaxf(mrow[r], mx);
      const float corr = __expf(mrow[r] - mnew);
      mrow[r] = mnew;
      float rsum = 0.0f;
      #pragma unroll
      for (int nt = 0; nt < 8; ++nt) {
        float p = __expf(s[nt][r] - mnew);
        s[nt][r] = p;
        rsum += p;
      }
      #pragma unroll
      for (int off = 1; off < 16; off <<= 1) rsum += __shfl_xor(rsum, off, 32);
      lrow[r] = lrow[r] * corr + rsum;
      #pragma unroll
      for (int ot = 0; ot < 4; ++ot) o[ot][r] *= corr;
    }

    // P -> wave-private LDS (reshape C-layout -> A-operand layout)
    #pragma unroll
    for (int nt = 0; nt < 8; ++nt)
      #pragma unroll
      for (int r = 0; r < 8; ++r)
        Pw[(rbase + r) * PSTR + nt * 16 + cidx] = (__bf16)s[nt][r];
    // same-wave LDS ordering: no barrier needed

    // O += P V  (4 k-steps of 32 over the 128 keys, 4 output d-tiles)
    #pragma unroll
    for (int ks = 0; ks < 4; ++ks) {
      v16bf pf = ld_frag(Pw, PSTR, 0, ks * 32);
      #pragma unroll
      for (int ot = 0; ot < 4; ++ot) {
        v16bf vf = ld_frag(Vt, VSTR, ot * 16, ks * 32);
        o[ot] = __builtin_amdgcn_wmma_f32_16x16x32_bf16(false, pf, false, vf,
                                                        (short)0, o[ot], false, false);
      }
    }
  }

  // finalize: O /= l, store bf16 to [B*N, H*D]
  #pragma unroll
  for (int r = 0; r < 8; ++r) {
    const float inv = 1.0f / lrow[r];
    const int row = q0 + rbase + r;
    const size_t orow = (size_t)(b * Nq + row);
    #pragma unroll
    for (int ot = 0; ot < 4; ++ot)
      out[orow * 1024 + h * D + ot * 16 + cidx] = (__bf16)(o[ot][r] * inv);
  }
}

// ---------------------------------------------------------------------------
extern "C" void kernel_launch(void* const* d_in, const int* in_sizes, int n_in,
                              void* d_out, int out_size, void* d_ws, size_t ws_size,
                              hipStream_t stream) {
  const float* x      = (const float*)d_in[0];  // [2,2048,1024]
  const float* w_qkv  = (const float*)d_in[1];  // [3072,1024]
  const float* w_proj = (const float*)d_in[2];  // [1024,1024]
  const float* b_proj = (const float*)d_in[3];  // [1024]

  constexpr int M = 4096, C = 1024, C3 = 3072;

  __bf16* xb     = (__bf16*)d_ws;
  __bf16* wqkvb  = xb     + (size_t)M * C;
  __bf16* wprojb = wqkvb  + (size_t)C3 * C;
  __bf16* qkvb   = wprojb + (size_t)C * C;
  __bf16* attnb  = qkvb   + (size_t)M * C3;   // total ws use: ~48 MB

  auto cvtGrid = [](int n) { return dim3((n / 4 + 255) / 256); };
  cvt_f32_to_bf16<<<cvtGrid(M * C),  256, 0, stream>>>(x,      xb,     M * C);
  cvt_f32_to_bf16<<<cvtGrid(C3 * C), 256, 0, stream>>>(w_qkv,  wqkvb,  C3 * C);
  cvt_f32_to_bf16<<<cvtGrid(C * C),  256, 0, stream>>>(w_proj, wprojb, C * C);

  // qkv = x @ w_qkv^T  -> bf16 [4096, 3072]
  gemm_wmma<true, false><<<dim3(C3 / 128, M / 128), 256, 0, stream>>>(
      xb, wqkvb, nullptr, qkvb, C, C3);

  // causal flash attention -> bf16 [4096, 1024]
  attn_wmma<<<dim3(16, 32), 256, 0, stream>>>(qkvb, attnb);

  // out = attn @ w_proj^T + b  -> fp32 d_out
  gemm_wmma<false, true><<<dim3(C / 128, M / 128), 256, 0, stream>>>(
      attnb, wprojb, b_proj, (float*)d_out, C, C);
}
